// TransformerBlock_55482387530294
// MI455X (gfx1250) — matrix operand order repaired and checked by reading, not compiled, not run on hardware
//
#include <hip/hip_runtime.h>
#include <hip/hip_bf16.h>
#include <math.h>

// ---------------------------------------------------------------------------
// Transformer block for MI455X (gfx1250, wave32, WMMA + Tensor Data Mover).
// GEMMs: v_wmma_f32_16x16x32_bf16, tiles staged by TDM (tensor_load_to_lds),
// double-buffered LDS, TENSORcnt synchronization.
// ---------------------------------------------------------------------------

typedef __bf16 bf16_t;
typedef __attribute__((ext_vector_type(16))) __bf16 v16bf;
typedef __attribute__((ext_vector_type(8)))  float  v8f;
typedef __attribute__((ext_vector_type(4)))  unsigned int u32x4;
typedef __attribute__((ext_vector_type(8)))  int i32x8;
typedef __attribute__((ext_vector_type(4)))  int i32x4;

union FragAB { v16bf v; uint4 q[2]; unsigned short us[16]; };

#define B_   2
#define S_   2048
#define D_   1024
#define H_   16
#define HD_  64
#define F_   4096
#define BS_  (B_*S_)

// ---------------------------------------------------------------------------
// Tensor Data Mover: stage a 128-row x 32-col bf16 tile of a row-major
// [rows, K] tensor into LDS (contiguous [128][32]).  D# packing per CDNA5
// ISA ch.8: group0 {count=1, lds_addr, global_addr, type=2}, group1
// {data_size=2B, tensor_dim0=K, tensor_dim1=rows, tile=32x128, stride0=K}.
// Groups 2/3 zero (2D tensor). Issued per-wave (EXEC ignored).
// ---------------------------------------------------------------------------
__device__ __forceinline__ unsigned lds_off(const void* p) {
  return (unsigned)(uintptr_t)p;   // LDS aperture: addr[31:0] = LDS byte offset
}

__device__ __forceinline__ void tdm_load_tile(unsigned lds_addr,
                                              const bf16_t* gbase_row,
                                              int K, int rows, int k0) {
  const unsigned long long ga = (unsigned long long)(uintptr_t)(gbase_row + k0);
  u32x4 g0;
  g0[0] = 1u;                                            // count=1 (valid load D#)
  g0[1] = lds_addr;                                      // lds_addr[31:0]
  g0[2] = (unsigned)(ga & 0xffffffffu);                  // global_addr[31:0]
  g0[3] = (unsigned)((ga >> 32) & 0x01ffffffu)           // global_addr[56:32]
        | 0x80000000u;                                   // type=2 ("image")
  i32x8 g1;
  g1[0] = (int)(1u << 16);                               // data_size=1 -> 2 bytes
  g1[1] = (int)(((unsigned)K & 0xffffu) << 16);          // tensor_dim0[15:0]
  g1[2] = (int)((((unsigned)K >> 16) & 0xffffu)          // tensor_dim0[31:16]
        | (((unsigned)rows & 0xffffu) << 16));           // tensor_dim1[15:0]
  g1[3] = (int)((((unsigned)rows >> 16) & 0xffffu)       // tensor_dim1[31:16]
        | (32u << 16));                                  // tile_dim0 = 32
  g1[4] = 128;                                           // tile_dim1 = 128
  g1[5] = K;                                             // tensor_dim0_stride lo
  g1[6] = 0;
  g1[7] = 0;
  i32x4 gz = {0, 0, 0, 0};
#if defined(__clang_major__) && (__clang_major__ >= 23)
  i32x8 gz8 = {0, 0, 0, 0, 0, 0, 0, 0};
  __builtin_amdgcn_tensor_load_to_lds(g0, g1, gz, gz, gz8, 0);
#else
  __builtin_amdgcn_tensor_load_to_lds(g0, g1, gz, gz, 0);
#endif
}

// ---------------------------------------------------------------------------
// Weight prep: f32 [Kdim,Ndim] -> bf16 transposed [Ndim,Kdim] (32x32 LDS tile).
// Makes every GEMM operand a row-major panel -> TDM-stageable, no LDS
// transpose in the hot loop.
// ---------------------------------------------------------------------------
__global__ __launch_bounds__(256) void cvtT_f32_bf16(const float* __restrict__ in,
                                                     bf16_t* __restrict__ out,
                                                     int Kdim, int Ndim) {
  __shared__ float tile[32][33];
  const int t = threadIdx.x;
  const int tx = t & 31, ty = t >> 5;
  const int n0 = blockIdx.x * 32, k0 = blockIdx.y * 32;
  #pragma unroll
  for (int i = 0; i < 32; i += 8)
    tile[ty + i][tx] = in[(size_t)(k0 + ty + i) * Ndim + n0 + tx];
  __syncthreads();
  #pragma unroll
  for (int i = 0; i < 32; i += 8)
    out[(size_t)(n0 + ty + i) * Kdim + k0 + tx] = (bf16_t)tile[tx][ty + i];
}

// ---------------------------------------------------------------------------
// LayerNorm: one block per row of D_=1024, 256 threads x 4 elements -> bf16.
// ---------------------------------------------------------------------------
struct bf4 { bf16_t x, y, z, w; };

__global__ __launch_bounds__(256) void ln_bf16(const float* __restrict__ x,
                                               const float* __restrict__ g,
                                               const float* __restrict__ bta,
                                               bf16_t* __restrict__ out) {
  const int row = blockIdx.x;
  const int t = threadIdx.x;
  const float4 v = ((const float4*)(x + (size_t)row * D_))[t];
  float s  = v.x + v.y + v.z + v.w;
  float ss = v.x*v.x + v.y*v.y + v.z*v.z + v.w*v.w;
  #pragma unroll
  for (int m = 16; m >= 1; m >>= 1) {
    s  += __shfl_xor(s,  m, 32);
    ss += __shfl_xor(ss, m, 32);
  }
  __shared__ float rs[8], rss[8];
  const int w = t >> 5, lane = t & 31;
  if (lane == 0) { rs[w] = s; rss[w] = ss; }
  __syncthreads();
  s = 0.f; ss = 0.f;
  #pragma unroll
  for (int i = 0; i < 8; ++i) { s += rs[i]; ss += rss[i]; }
  const float mean = s * (1.0f / D_);
  const float var  = ss * (1.0f / D_) - mean * mean;
  const float rstd = rsqrtf(var + 1e-5f);
  const float4 gv = ((const float4*)g)[t];
  const float4 bv = ((const float4*)bta)[t];
  bf4 o;
  o.x = (bf16_t)((v.x - mean) * rstd * gv.x + bv.x);
  o.y = (bf16_t)((v.y - mean) * rstd * gv.y + bv.y);
  o.z = (bf16_t)((v.z - mean) * rstd * gv.z + bv.z);
  o.w = (bf16_t)((v.w - mean) * rstd * gv.w + bv.w);
  ((bf4*)(out + (size_t)row * D_))[t] = o;
}

// ---------------------------------------------------------------------------
// Tiled WMMA GEMM: C[M,N] = epilogue(A[M,K] @ Bt[N,K]^T).
// Block tile 128x128, BK=32, 8 waves (each 64x32). Both operand tiles are
// row-major 128x32 panels staged by TDM into double-buffered LDS; wave 0
// drives the DMA, S_WAIT_TENSORCNT + barrier publish each buffer.
// Fragment loads are contiguous ds_load_b128 per the CDNA5 WMMA layouts
// (A: lane=m, k-halves {base, 16+base}; B: lane=n, k-halves {base, 16+base}).
// ---------------------------------------------------------------------------
enum { EPI_Q, EPI_K, EPI_V, EPI_O, EPI_F1, EPI_F2 };

template<int EPI>
__global__ __launch_bounds__(256) void gemm_bf16_k(
    const bf16_t* __restrict__ A, const bf16_t* __restrict__ Bt,
    int M, int N, int K,
    bf16_t* __restrict__ outB, float* __restrict__ outF,
    const float* __restrict__ resid)
{
  __shared__ __align__(16) unsigned short As[2][128 * 32];
  __shared__ __align__(16) unsigned short Bs[2][128 * 32];

  const int t    = threadIdx.x;
  const int lane = t & 31, w = t >> 5;
  const int wm   = w & 1,  wn = w >> 1;
  const int m0   = blockIdx.y * 128, n0 = blockIdx.x * 128;
  const int hi   = lane >> 4, l16 = lane & 15;

  const bf16_t* Apanel = A  + (size_t)m0 * K;   // 128 rows of A
  const bf16_t* Bpanel = Bt + (size_t)n0 * K;   // 128 rows of B^T

  v8f acc[4][2] = {};
  const int nsteps = K >> 5;

  if (w == 0) {
    tdm_load_tile(lds_off(&As[0][0]), Apanel, K, M, 0);
    tdm_load_tile(lds_off(&Bs[0][0]), Bpanel, K, N, 0);
  }

  for (int step = 0; step < nsteps; ++step) {
    const int cur = step & 1;
    if (w == 0) __builtin_amdgcn_s_wait_tensorcnt(0);  // buffer `cur` landed
    __syncthreads();
    if (w == 0 && step + 1 < nsteps) {                 // prefetch next buffer
      const int k0n = (step + 1) << 5;
      tdm_load_tile(lds_off(&As[cur ^ 1][0]), Apanel, K, M, k0n);
      tdm_load_tile(lds_off(&Bs[cur ^ 1][0]), Bpanel, K, N, k0n);
    }

    const unsigned short* Ab = &As[cur][0];
    const unsigned short* Bb = &Bs[cur][0];

    FragAB af[4], bfr[2];
    #pragma unroll
    for (int i = 0; i < 4; ++i) {
      const int m = wm * 64 + i * 16 + l16;
      const int base = hi * 8;
      af[i].q[0] = *(const uint4*)(&Ab[m * 32 + base]);
      af[i].q[1] = *(const uint4*)(&Ab[m * 32 + 16 + base]);
    }
    #pragma unroll
    for (int j = 0; j < 2; ++j) {
      const int n = wn * 32 + j * 16 + l16;
      const int kb = hi * 8;
      bfr[j].q[0] = *(const uint4*)(&Bb[n * 32 + kb]);
      bfr[j].q[1] = *(const uint4*)(&Bb[n * 32 + kb + 16]);
    }
    #pragma unroll
    for (int i = 0; i < 4; ++i)
      #pragma unroll
      for (int j = 0; j < 2; ++j)
        acc[i][j] = __builtin_amdgcn_wmma_f32_16x16x32_bf16(
            false, af[i].v, false, bfr[j].v, (short)0, acc[i][j], false, false);
  }

  // ---- epilogue ----
  #pragma unroll
  for (int i = 0; i < 4; ++i)
    #pragma unroll
    for (int j = 0; j < 2; ++j)
      #pragma unroll
      for (int r = 0; r < 8; ++r) {
        const int m = m0 + wm * 64 + i * 16 + r + hi * 8;
        const int n = n0 + wn * 32 + j * 16 + l16;
        const float val = acc[i][j][r];
        if constexpr (EPI == EPI_Q || EPI == EPI_K) {
          const int b = m >> 11, s = m & (S_ - 1);
          const int h = n >> 6,  hd = n & (HD_ - 1);
          const size_t off = ((size_t)((b * H_ + h) * S_ + s)) * HD_ + hd;
          outB[off] = (bf16_t)val;
          if constexpr (EPI == EPI_K) outF[off] = val;
        } else if constexpr (EPI == EPI_V) {
          const int b = m >> 11, s = m & (S_ - 1);
          const int h = n >> 6,  hd = n & (HD_ - 1);
          const size_t offT = ((size_t)((b * H_ + h) * HD_ + hd)) * S_ + s;
          const size_t off  = ((size_t)((b * H_ + h) * S_ + s)) * HD_ + hd;
          outB[offT] = (bf16_t)val;          // V^T [b][h][hd][s] for attention
          outF[off]  = val;                  // f32 output (b,h,s,hd)
        } else if constexpr (EPI == EPI_O || EPI == EPI_F2) {
          const size_t off = (size_t)m * N + n;
          outF[off] = resid[off] + val;      // residual add
        } else {                             // EPI_F1: exact GELU -> bf16
          const size_t off = (size_t)m * N + n;
          const float gl = 0.5f * val * (1.0f + erff(val * 0.70710678118654752f));
          outB[off] = (bf16_t)gl;
        }
      }
}

// ---------------------------------------------------------------------------
// Flash attention, causal. Grid (B*H, S/128), 8 independent waves per block;
// each wave owns 16 query rows, streams K/V in 32-wide chunks straight from
// global as 32B fragments (K stored [s][hd], V stored transposed [hd][s]).
// P transposes C-layout -> A-layout through a per-wave LDS strip guarded by
// an explicit s_wait_dscnt (wave-internal, no block barrier).
// ---------------------------------------------------------------------------
__global__ __launch_bounds__(256) void attn_k(
    const bf16_t* __restrict__ Q, const bf16_t* __restrict__ Kb,
    const bf16_t* __restrict__ Vt, bf16_t* __restrict__ ctx)
{
  __shared__ __align__(16) unsigned short Plds[8][16 * 32];

  const int t    = threadIdx.x;
  const int lane = t & 31, w = t >> 5;
  const int hi   = lane >> 4, l16 = lane & 15;
  const int bh   = blockIdx.x;
  const int b    = bh >> 4, h = bh & 15;
  const int q0   = blockIdx.y * 128 + w * 16;

  const bf16_t* qbase = Q  + ((size_t)bh * S_) * HD_;
  const bf16_t* kbase = Kb + ((size_t)bh * S_) * HD_;
  const bf16_t* vbase = Vt + ((size_t)bh * HD_) * S_;

  FragAB aq[2];
  {
    const bf16_t* qp = qbase + (size_t)(q0 + l16) * HD_;
    #pragma unroll
    for (int ks = 0; ks < 2; ++ks) {
      const int base = ks * 32 + hi * 8;
      aq[ks].q[0] = *(const uint4*)(qp + base);
      aq[ks].q[1] = *(const uint4*)(qp + base + 16);
    }
  }

  v8f accc[4] = {};
  float rmax[8], rsum[8];
  #pragma unroll
  for (int r = 0; r < 8; ++r) { rmax[r] = -1e30f; rsum[r] = 0.f; }

  unsigned short* P = &Plds[w][0];

  for (int kb = 0; kb < q0 + 16; kb += 32) {
    v8f sc[2] = {};
    #pragma unroll
    for (int hh = 0; hh < 2; ++hh) {
      FragAB bk[2];
      const bf16_t* kp = kbase + (size_t)(kb + hh * 16 + l16) * HD_;
      #pragma unroll
      for (int ks = 0; ks < 2; ++ks) {
        const int base = ks * 32 + hi * 8;
        bk[ks].q[0] = *(const uint4*)(kp + base);
        bk[ks].q[1] = *(const uint4*)(kp + base + 16);
      }
      sc[hh] = __builtin_amdgcn_wmma_f32_16x16x32_bf16(
          false, aq[0].v, false, bk[0].v, (short)0, sc[hh], false, false);
      sc[hh] = __builtin_amdgcn_wmma_f32_16x16x32_bf16(
          false, aq[1].v, false, bk[1].v, (short)0, sc[hh], false, false);
    }

    const bool needmask = (kb + 31 > q0);
    #pragma unroll
    for (int r = 0; r < 8; ++r) {
      float s0 = sc[0][r] * 0.125f;
      float s1 = sc[1][r] * 0.125f;
      if (needmask) {
        const int qrow = q0 + r + hi * 8;
        if (kb + l16      > qrow) s0 = s0 - 1e9f;
        if (kb + 16 + l16 > qrow) s1 = s1 - 1e9f;
      }
      float mx = fmaxf(s0, s1);
      #pragma unroll
      for (int m = 8; m >= 1; m >>= 1) mx = fmaxf(mx, __shfl_xor(mx, m, 32));
      const float nm   = fmaxf(rmax[r], mx);
      const float corr = __expf(rmax[r] - nm);
      rmax[r] = nm;
      const float p0 = __expf(s0 - nm);
      const float p1 = __expf(s1 - nm);
      float ps = p0 + p1;
      #pragma unroll
      for (int m = 8; m >= 1; m >>= 1) ps += __shfl_xor(ps, m, 32);
      rsum[r] = rsum[r] * corr + ps;
      #pragma unroll
      for (int nn = 0; nn < 4; ++nn) accc[nn][r] *= corr;
      const int mrow = r + hi * 8;
      const bf16_t pb0 = (bf16_t)p0, pb1 = (bf16_t)p1;
      P[mrow * 32 + l16]      = __builtin_bit_cast(unsigned short, pb0);
      P[mrow * 32 + 16 + l16] = __builtin_bit_cast(unsigned short, pb1);
    }
    asm volatile("s_wait_dscnt 0x0" ::: "memory");

    FragAB ap;
    {
      const int base = hi * 8;
      ap.q[0] = *(const uint4*)(&P[l16 * 32 + base]);
      ap.q[1] = *(const uint4*)(&P[l16 * 32 + 16 + base]);
    }
    #pragma unroll
    for (int nn = 0; nn < 4; ++nn) {
      FragAB bv;
      const bf16_t* vp = vbase + (size_t)(nn * 16 + l16) * S_ + kb + hi * 8;
      bv.q[0] = *(const uint4*)(vp);
      bv.q[1] = *(const uint4*)(vp + 16);
      accc[nn] = __builtin_amdgcn_wmma_f32_16x16x32_bf16(
          false, ap.v, false, bv.v, (short)0, accc[nn], false, false);
    }
  }

  #pragma unroll
  for (int r = 0; r < 8; ++r) {
    const int mrow = r + hi * 8;
    const float inv = 1.0f / rsum[r];
    const size_t rowoff = ((size_t)b * S_ + (q0 + mrow)) * D_ + h * HD_;
    #pragma unroll
    for (int nn = 0; nn < 4; ++nn)
      ctx[rowoff + nn * 16 + l16] = (bf16_t)(accc[nn][r] * inv);
  }
}

// ---------------------------------------------------------------------------
// Host orchestration
// ---------------------------------------------------------------------------
extern "C" void kernel_launch(void* const* d_in, const int* in_sizes, int n_in,
                              void* d_out, int out_size, void* d_ws, size_t ws_size,
                              hipStream_t stream) {
  const float* x    = (const float*)d_in[0];   // hidden_states (B,S,D)
  // d_in[1] attention_mask: causal mask reproduced analytically in attn_k
  const float* ln1g = (const float*)d_in[2];
  const float* ln1b = (const float*)d_in[3];
  const float* Wq   = (const float*)d_in[4];
  const float* Wk   = (const float*)d_in[5];
  const float* Wv   = (const float*)d_in[6];
  const float* Wo   = (const float*)d_in[7];
  const float* ln2g = (const float*)d_in[8];
  const float* ln2b = (const float*)d_in[9];
  const float* W1   = (const float*)d_in[10];
  const float* W2   = (const float*)d_in[11];

  const size_t DD  = (size_t)D_ * D_;
  const size_t DF  = (size_t)D_ * F_;
  const size_t BSD = (size_t)BS_ * D_;
  const size_t BSF = (size_t)BS_ * F_;

  float* out   = (float*)d_out;
  float* k_out = out + BSD;
  float* v_out = out + 2 * BSD;

  char* ws = (char*)d_ws;
  size_t off = 0;
  auto take = [&](size_t bytes) {
    char* p = ws + off;
    off = (off + bytes + 255) & ~(size_t)255;
    return p;
  };
  bf16_t* wqT  = (bf16_t*)take(DD * 2);   // Wq^T [D][D]
  bf16_t* wkT  = (bf16_t*)take(DD * 2);
  bf16_t* wvT  = (bf16_t*)take(DD * 2);
  bf16_t* woT  = (bf16_t*)take(DD * 2);
  bf16_t* w1T  = (bf16_t*)take(DF * 2);   // W1^T [F][D]
  bf16_t* w2T  = (bf16_t*)take(DF * 2);   // W2^T [D][F]
  bf16_t* h1   = (bf16_t*)take(BSD * 2);  // LN1 out
  bf16_t* qb   = (bf16_t*)take(BSD * 2);  // Q   (b,h,s,hd)
  bf16_t* kbuf = (bf16_t*)take(BSD * 2);  // K   (b,h,s,hd)
  bf16_t* vtb  = (bf16_t*)take(BSD * 2);  // V^T (b,h,hd,s)
  bf16_t* ctxb = (bf16_t*)take(BSD * 2);  // attention context (b,s,d)
  float*  x1   = (float*) take(BSD * 4);  // x + ctx@Wo
  bf16_t* h2   = (bf16_t*)take(BSD * 2);  // LN2 out
  bf16_t* ffn1 = (bf16_t*)take(BSF * 2);  // gelu(h2@W1)
  (void)ws_size; (void)in_sizes; (void)n_in; (void)out_size;

  // weights: f32 [K,N] -> bf16 transposed [N,K]
  cvtT_f32_bf16<<<dim3(D_/32, D_/32), 256, 0, stream>>>(Wq, wqT, D_, D_);
  cvtT_f32_bf16<<<dim3(D_/32, D_/32), 256, 0, stream>>>(Wk, wkT, D_, D_);
  cvtT_f32_bf16<<<dim3(D_/32, D_/32), 256, 0, stream>>>(Wv, wvT, D_, D_);
  cvtT_f32_bf16<<<dim3(D_/32, D_/32), 256, 0, stream>>>(Wo, woT, D_, D_);
  cvtT_f32_bf16<<<dim3(F_/32, D_/32), 256, 0, stream>>>(W1, w1T, D_, F_);
  cvtT_f32_bf16<<<dim3(D_/32, F_/32), 256, 0, stream>>>(W2, w2T, F_, D_);

  ln_bf16<<<BS_, 256, 0, stream>>>(x, ln1g, ln1b, h1);

  const dim3 gD(D_ / 128, BS_ / 128);   // (8, 32)
  gemm_bf16_k<EPI_Q><<<gD, 256, 0, stream>>>(h1, wqT, BS_, D_, D_, qb,   nullptr, nullptr);
  gemm_bf16_k<EPI_K><<<gD, 256, 0, stream>>>(h1, wkT, BS_, D_, D_, kbuf, k_out,   nullptr);
  gemm_bf16_k<EPI_V><<<gD, 256, 0, stream>>>(h1, wvT, BS_, D_, D_, vtb,  v_out,   nullptr);

  attn_k<<<dim3(B_ * H_, S_ / 128), 256, 0, stream>>>(qb, kbuf, vtb, ctxb);

  gemm_bf16_k<EPI_O><<<gD, 256, 0, stream>>>(ctxb, woT, BS_, D_, D_, nullptr, x1, x);

  ln_bf16<<<BS_, 256, 0, stream>>>(x1, ln2g, ln2b, h2);

  gemm_bf16_k<EPI_F1><<<dim3(F_ / 128, BS_ / 128), 256, 0, stream>>>(
      h2, w1T, BS_, F_, D_, ffn1, nullptr, nullptr);
  gemm_bf16_k<EPI_F2><<<dim3(D_ / 128, BS_ / 128), 256, 0, stream>>>(
      ffn1, w2T, BS_, D_, F_, nullptr, out, x1);
}
